// ENN_S2S_48902497632443
// MI455X (gfx1250) — matrix operand order repaired and checked by reading, not compiled, not run on hardware
//
#include <hip/hip_runtime.h>
#include <hip/hip_bf16.h>

#define NN 131072   // nodes
#define NE 524288   // edges
#define NG 4096     // graphs
#define DH 256      // hidden dim

typedef __hip_bfloat16 bf16;
typedef __attribute__((ext_vector_type(16))) __bf16 v16bf;
typedef __attribute__((ext_vector_type(8)))  float  v8f;

union ABFrag { unsigned int u[8]; uint4 q[2]; v16bf v; };

// Async global -> LDS copy (CDNA5 async path, ASYNCcnt tracked).
// LDS operand is the 32-bit LDS byte offset (low 32 bits of generic address).
__device__ __forceinline__ void async_ld_b128(void* lds, const void* g) {
    unsigned l = (unsigned)(unsigned long long)lds;
    asm volatile("global_load_async_to_lds_b128 %0, %1, off"
                 :: "v"(l), "v"(g) : "memory");
}
__device__ __forceinline__ void wait_async4() {
    asm volatile("s_wait_asynccnt 0x4" ::: "memory");
}
__device__ __forceinline__ void wait_async0() {
    asm volatile("s_wait_asynccnt 0x0" ::: "memory");
}

// ---------------------------------------------------------------------------
// fp32 -> bf16 conversion
// ---------------------------------------------------------------------------
__global__ __launch_bounds__(256) void f2b_kernel(const float* __restrict__ in,
                                                  bf16* __restrict__ out, size_t n) {
    size_t i = (size_t)blockIdx.x * 256 + threadIdx.x;
    if (i < n) out[i] = __float2bfloat16(in[i]);
}

// fp32 [K,N] row-major -> bf16 [N,K] row-major (transpose + convert)
__global__ __launch_bounds__(256) void f2bT_kernel(const float* __restrict__ in,
                                                   bf16* __restrict__ out,
                                                   int K, int N) {
    size_t i = (size_t)blockIdx.x * 256 + threadIdx.x;
    if (i < (size_t)K * N) {
        int k = (int)(i / N), n = (int)(i % N);
        out[(size_t)n * K + k] = __float2bfloat16(in[i]);
    }
}

// ---------------------------------------------------------------------------
// Atom embedding: x = sum_f atom_emb[f][x_atoms[:,f]]; also init GRU hidden
// ---------------------------------------------------------------------------
__global__ __launch_bounds__(256) void embed_kernel(
    const int* __restrict__ x_atoms, const float* __restrict__ atom_emb,
    float* __restrict__ xf, float* __restrict__ hx0f, float* __restrict__ hx1f,
    bf16* __restrict__ hxb0, bf16* __restrict__ hxb1) {
    int n = (int)(((size_t)blockIdx.x * 256 + threadIdx.x) >> 5);
    int lane = threadIdx.x & 31;
    if (n >= NN) return;
    int fv[9];
    #pragma unroll
    for (int f = 0; f < 9; ++f) fv[f] = x_atoms[(size_t)n * 9 + f];
    for (int d = lane; d < DH; d += 32) {
        float s = 0.0f;
        #pragma unroll
        for (int f = 0; f < 9; ++f)
            s += atom_emb[((size_t)f * 120 + fv[f]) * DH + d];
        size_t o = (size_t)n * DH + d;
        xf[o] = s; hx0f[o] = s; hx1f[o] = s;
        bf16 b = __float2bfloat16(s);
        hxb0[o] = b; hxb1[o] = b;
    }
}

// ---------------------------------------------------------------------------
// Edge messages: agg[dst] += relu(x[src] + bond_embed(edge)); wave per edge
// ---------------------------------------------------------------------------
__global__ __launch_bounds__(256) void message_kernel(
    const float* __restrict__ xf, const int* __restrict__ ei,
    const int* __restrict__ ea, const float* __restrict__ bond_emb,
    float* __restrict__ agg) {
    int e = (int)(((size_t)blockIdx.x * 256 + threadIdx.x) >> 5);
    int lane = threadIdx.x & 31;
    if (e >= NE) return;
    int src = ei[e], dst = ei[NE + e];
    const float* b0 = bond_emb + (size_t)(0 * 6 + ea[(size_t)e * 3 + 0]) * DH;
    const float* b1 = bond_emb + (size_t)(1 * 6 + ea[(size_t)e * 3 + 1]) * DH;
    const float* b2 = bond_emb + (size_t)(2 * 6 + ea[(size_t)e * 3 + 2]) * DH;
    const float* xs = xf + (size_t)src * DH;
    float* ag = agg + (size_t)dst * DH;
    for (int d = lane; d < DH; d += 32) {
        float m = xs[d] + b0[d] + b1[d] + b2[d];
        unsafeAtomicAdd(&ag[d], fmaxf(m, 0.0f));
    }
}

// ---------------------------------------------------------------------------
// bf16 WMMA GEMM: C[M,N] = A[M,K] @ B^T (+bias), B stored [N,K] row-major.
// EPI: 0 = fp32 out, 1 = relu -> bf16 out, 2 = bf16 out
// Tile 128x128x32, 256 threads (8 waves), wave tile 64x32 (4x2 WMMA frags).
// Double-buffered LDS, async global->LDS prefetch, peeled final K-step so the
// steady-state inner loop is branch-free.
// ---------------------------------------------------------------------------
template<int EPI>
__global__ __launch_bounds__(256) void gemm_kernel(
    const bf16* __restrict__ A, const bf16* __restrict__ B,
    const float* __restrict__ bias,
    float* __restrict__ Cf, bf16* __restrict__ Cb,
    int M, int N, int K) {
    __shared__ __align__(16) unsigned short sA[2][128 * 32];  // [m][k]
    __shared__ __align__(16) unsigned short sB[2][128 * 32];  // [n][k]

    const int tid  = threadIdx.x;
    const int wid  = tid >> 5;
    const int lane = tid & 31;
    const int wm = wid >> 2;       // 0..1 : wave row
    const int wn = wid & 3;        // 0..3 : wave col
    const int hfv = lane >> 4;     // lane half (0/1)
    const int ln  = lane & 15;
    const int m0 = blockIdx.y * 128;
    const int n0 = blockIdx.x * 128;

    // Per-thread tile chunks: c0 = tid, c1 = tid+256 of 512 b128 chunks
    const int mA0 = tid >> 2,           kA0 = (tid & 3) << 3;
    const int mA1 = (tid + 256) >> 2,   kA1 = kA0;   // (c+256)&3 == c&3

    v8f acc[4][2];
    #pragma unroll
    for (int i = 0; i < 4; ++i)
        #pragma unroll
        for (int j = 0; j < 2; ++j)
            #pragma unroll
            for (int v = 0; v < 8; ++v) acc[i][j][v] = 0.0f;

    // Issue one tile's async loads: 2x A + 2x B = ASYNCcnt +4 per wave
    auto issue_tile = [&](int buf, int kk) {
        async_ld_b128(&sA[buf][mA0 * 32 + kA0],
                      A + (size_t)(m0 + mA0) * K + kk + kA0);
        async_ld_b128(&sA[buf][mA1 * 32 + kA1],
                      A + (size_t)(m0 + mA1) * K + kk + kA1);
        async_ld_b128(&sB[buf][mA0 * 32 + kA0],
                      B + (size_t)(n0 + mA0) * K + kk + kA0);
        async_ld_b128(&sB[buf][mA1 * 32 + kA1],
                      B + (size_t)(n0 + mA1) * K + kk + kA1);
    };

    // Fragments per ISA layout: A 16x32 (lanes 0-15 K=0..15 pattern, lanes
    // 16-31 shifted by 8 / 24); B 32x16 (same K pattern as A); then 8 WMMAs.
    auto compute_tile = [&](int buf) {
        ABFrag afr[4], bfr[2];
        #pragma unroll
        for (int i = 0; i < 4; ++i) {
            int r = wm * 64 + i * 16 + ln;
            afr[i].q[0] = *(const uint4*)&sA[buf][r * 32 + hfv * 8];
            afr[i].q[1] = *(const uint4*)&sA[buf][r * 32 + 16 + hfv * 8];
        }
        #pragma unroll
        for (int j = 0; j < 2; ++j) {
            int cc = wn * 32 + j * 16 + ln;
            bfr[j].q[0] = *(const uint4*)&sB[buf][cc * 32 + hfv * 8];
            bfr[j].q[1] = *(const uint4*)&sB[buf][cc * 32 + 16 + hfv * 8];
        }
        #pragma unroll
        for (int i = 0; i < 4; ++i)
            #pragma unroll
            for (int j = 0; j < 2; ++j)
                acc[i][j] = __builtin_amdgcn_wmma_f32_16x16x32_bf16(
                    false, afr[i].v, false, bfr[j].v,
                    (short)0, acc[i][j], false, false);
    };

    issue_tile(0, 0);

    // Steady state: always prefetch next tile, branch-free body.
    int k0 = 0;
    for (; k0 + 32 < K; k0 += 32) {
        const int buf = (k0 >> 5) & 1;
        issue_tile(buf ^ 1, k0 + 32);
        wait_async4();                 // retire current tile's 4 ops
        __syncthreads();
        compute_tile(buf);
        __syncthreads();               // buffer reusable after this
    }
    // Peeled final tile: nothing left to prefetch.
    wait_async0();
    __syncthreads();
    compute_tile((k0 >> 5) & 1);

    // Epilogue: C layout -> row = half*8+v, col = lane%16
    #pragma unroll
    for (int j = 0; j < 2; ++j) {
        int col = n0 + wn * 32 + j * 16 + ln;
        float bv = bias ? bias[col] : 0.0f;
        #pragma unroll
        for (int i = 0; i < 4; ++i) {
            #pragma unroll
            for (int v = 0; v < 8; ++v) {
                int row = m0 + wm * 64 + i * 16 + hfv * 8 + v;
                float val = acc[i][j][v] + bv;
                if (EPI == 1) val = fmaxf(val, 0.0f);
                size_t o = (size_t)row * N + col;
                if (EPI == 0) Cf[o] = val;
                else          Cb[o] = __float2bfloat16(val);
            }
        }
    }
}

// ---------------------------------------------------------------------------
// GRU gate fusion (torch order r,z,n). In-place hidden update + bf16 copy.
// ---------------------------------------------------------------------------
__global__ __launch_bounds__(256) void gru_gate_kernel(
    const float* __restrict__ gi, const float* __restrict__ gh,
    float* __restrict__ h, bf16* __restrict__ hb, float* __restrict__ x_out) {
    size_t idx = (size_t)blockIdx.x * 256 + threadIdx.x;   // NN*DH exact
    int n = (int)(idx >> 8), d = (int)(idx & 255);
    const float* gin = gi + (size_t)n * 768;
    const float* ghn = gh + (size_t)n * 768;
    float r  = 1.0f / (1.0f + __expf(-(gin[d]       + ghn[d])));
    float z  = 1.0f / (1.0f + __expf(-(gin[256 + d] + ghn[256 + d])));
    float nn = tanhf(gin[512 + d] + r * ghn[512 + d]);
    float hp = h[idx];
    float hn = (1.0f - z) * nn + z * hp;
    h[idx] = hn;
    hb[idx] = __float2bfloat16(hn);
    if (x_out) x_out[idx] = hn;
}

// ---------------------------------------------------------------------------
// LSTM cell fusion (torch order i,f,g,o); g = g1 + g2
// ---------------------------------------------------------------------------
__global__ __launch_bounds__(256) void lstm_gate_kernel(
    const float* __restrict__ g1, const float* __restrict__ g2,
    float* __restrict__ c, float* __restrict__ h, bf16* __restrict__ hb) {
    size_t idx = (size_t)blockIdx.x * 256 + threadIdx.x;   // NG*DH exact
    int n = (int)(idx >> 8), d = (int)(idx & 255);
    const float* a = g1 + (size_t)n * 1024;
    const float* b = g2 + (size_t)n * 1024;
    float ig = a[d]        + b[d];
    float fg = a[256 + d]  + b[256 + d];
    float gg = a[512 + d]  + b[512 + d];
    float og = a[768 + d]  + b[768 + d];
    float cn = (1.0f / (1.0f + __expf(-fg))) * c[idx]
             + (1.0f / (1.0f + __expf(-ig))) * tanhf(gg);
    float hn = (1.0f / (1.0f + __expf(-og))) * tanhf(cn);
    c[idx] = cn; h[idx] = hn; hb[idx] = __float2bfloat16(hn);
}

// ---------------------------------------------------------------------------
// Graph segment bookkeeping (batch is sorted)
// ---------------------------------------------------------------------------
__global__ __launch_bounds__(256) void count_kernel(const int* __restrict__ batch,
                                                    int* __restrict__ cnt) {
    int i = blockIdx.x * 256 + threadIdx.x;
    if (i < NN) atomicAdd(&cnt[batch[i]], 1);
}

__global__ void scan_kernel(const int* __restrict__ cnt, int* __restrict__ off) {
    if (blockIdx.x == 0 && threadIdx.x == 0) {
        int acc = 0; off[0] = 0;
        for (int g = 0; g < NG; ++g) { acc += cnt[g]; off[g + 1] = acc; }
    }
}

// prod[i] = dot(q[batch[i]], x[i]); wave per node
__global__ __launch_bounds__(256) void prod_kernel(
    const float* __restrict__ q, const float* __restrict__ xf,
    const int* __restrict__ batch, float* __restrict__ prod) {
    int n = (int)(((size_t)blockIdx.x * 256 + threadIdx.x) >> 5);
    int lane = threadIdx.x & 31;
    if (n >= NN) return;
    const float* qv = q + (size_t)batch[n] * DH;
    const float* xv = xf + (size_t)n * DH;
    float s = 0.0f;
    for (int d = lane; d < DH; d += 32) s += qv[d] * xv[d];
    #pragma unroll
    for (int o = 16; o > 0; o >>= 1) s += __shfl_down(s, o);
    if (lane == 0) prod[n] = s;
}

// Block-per-graph softmax attention + q_star = [q, att_out] (fp32 + bf16)
__global__ __launch_bounds__(256) void attn_kernel(
    const int* __restrict__ off, const float* __restrict__ prod,
    const float* __restrict__ xf, const float* __restrict__ q,
    float* __restrict__ qstar, bf16* __restrict__ qstarb) {
    __shared__ float red[256];
    int g = blockIdx.x, tid = threadIdx.x;
    int s0 = off[g], s1 = off[g + 1];
    float m = -3.4e38f;
    for (int i = s0 + tid; i < s1; i += 256) m = fmaxf(m, prod[i]);
    red[tid] = m; __syncthreads();
    for (int s = 128; s > 0; s >>= 1) {
        if (tid < s) red[tid] = fmaxf(red[tid], red[tid + s]);
        __syncthreads();
    }
    m = red[0]; __syncthreads();
    float sum = 0.0f;
    for (int i = s0 + tid; i < s1; i += 256) sum += __expf(prod[i] - m);
    red[tid] = sum; __syncthreads();
    for (int s = 128; s > 0; s >>= 1) {
        if (tid < s) red[tid] += red[tid + s];
        __syncthreads();
    }
    float inv = 1.0f / (red[0] + 1e-10f);
    float acc = 0.0f;
    for (int i = s0; i < s1; ++i)
        acc += __expf(prod[i] - m) * inv * xf[(size_t)i * DH + tid];
    float qv = q[(size_t)g * DH + tid];
    qstar[(size_t)g * 512 + tid]       = qv;
    qstar[(size_t)g * 512 + 256 + tid] = acc;
    qstarb[(size_t)g * 512 + tid]       = __float2bfloat16(qv);
    qstarb[(size_t)g * 512 + 256 + tid] = __float2bfloat16(acc);
}

// out[g] = dot(q_star[g], fc_w) + fc_b; wave per graph
__global__ __launch_bounds__(256) void final_kernel(
    const float* __restrict__ qstar, const float* __restrict__ fc_w,
    const float* __restrict__ fc_b, float* __restrict__ out) {
    int g = (int)(((size_t)blockIdx.x * 256 + threadIdx.x) >> 5);
    int lane = threadIdx.x & 31;
    if (g >= NG) return;
    const float* qv = qstar + (size_t)g * 512;
    float s = 0.0f;
    for (int d = lane; d < 512; d += 32) s += qv[d] * fc_w[d];
    #pragma unroll
    for (int o = 16; o > 0; o >>= 1) s += __shfl_down(s, o);
    if (lane == 0) out[g] = s + fc_b[0];
}

// ---------------------------------------------------------------------------
extern "C" void kernel_launch(void* const* d_in, const int* in_sizes, int n_in,
                              void* d_out, int out_size, void* d_ws, size_t ws_size,
                              hipStream_t stream) {
    const int*   x_atoms  = (const int*)d_in[0];
    const int*   eidx     = (const int*)d_in[1];
    const int*   eattr    = (const int*)d_in[2];
    const int*   batch    = (const int*)d_in[3];
    const float* atom_emb = (const float*)d_in[4];
    const float* bond_emb = (const float*)d_in[5];
    const float* gin_w1   = (const float*)d_in[6];
    const float* gin_b1   = (const float*)d_in[7];
    const float* gin_w2   = (const float*)d_in[8];
    const float* gin_b2   = (const float*)d_in[9];
    const float* gru_wih  = (const float*)d_in[10];
    const float* gru_whh  = (const float*)d_in[11];
    const float* gru_bih  = (const float*)d_in[12];
    const float* gru_bhh  = (const float*)d_in[13];
    const float* lstm_wih = (const float*)d_in[14];
    const float* lstm_whh = (const float*)d_in[15];
    const float* lstm_bih = (const float*)d_in[16];
    const float* lstm_bhh = (const float*)d_in[17];
    const float* fc_w     = (const float*)d_in[18];
    const float* fc_b     = (const float*)d_in[19];
    float* outp = (float*)d_out;

    char* base = (char*)d_ws;
    size_t cur = 0;
    auto alloc = [&](size_t bytes) -> void* {
        void* p = base + cur;
        cur += (bytes + 255) & ~(size_t)255;
        return p;
    };
    float* xf   = (float*)alloc((size_t)NN * DH * 4);
    float* hx0f = (float*)alloc((size_t)NN * DH * 4);
    float* hx1f = (float*)alloc((size_t)NN * DH * 4);
    bf16*  hxb0 = (bf16*)alloc((size_t)NN * DH * 2);
    bf16*  hxb1 = (bf16*)alloc((size_t)NN * DH * 2);
    float* aggf = (float*)alloc((size_t)NN * DH * 4);
    bf16*  hbf  = (bf16*)alloc((size_t)NN * DH * 2);
    bf16*  tbf  = (bf16*)alloc((size_t)NN * 512 * 2);
    bf16*  xcb  = (bf16*)alloc((size_t)NN * DH * 2);
    float* gif  = (float*)alloc((size_t)NN * 768 * 4);
    float* ghf  = (float*)alloc((size_t)NN * 768 * 4);
    float* hsf  = (float*)alloc((size_t)NG * DH * 4);
    bf16*  hsb  = (bf16*)alloc((size_t)NG * DH * 2);
    float* csf  = (float*)alloc((size_t)NG * DH * 4);
    float* qsf  = (float*)alloc((size_t)NG * 512 * 4);
    bf16*  qsb  = (bf16*)alloc((size_t)NG * 512 * 2);
    float* g1f  = (float*)alloc((size_t)NG * 1024 * 4);
    float* g2f  = (float*)alloc((size_t)NG * 1024 * 4);
    float* prodv = (float*)alloc((size_t)NN * 4);
    int*   cnt  = (int*)alloc((size_t)NG * 4);
    int*   off  = (int*)alloc((size_t)(NG + 1) * 4);
    bf16* w1tb  = (bf16*)alloc((size_t)512 * 256 * 2);   // gin_w1^T [512,256]
    bf16* w2tb  = (bf16*)alloc((size_t)256 * 512 * 2);   // gin_w2^T [256,512]
    bf16* wihb  = (bf16*)alloc((size_t)2 * 768 * 256 * 2);
    bf16* whhb  = (bf16*)alloc((size_t)2 * 768 * 256 * 2);
    bf16* lwihb = (bf16*)alloc((size_t)1024 * 512 * 2);
    bf16* lwhhb = (bf16*)alloc((size_t)1024 * 256 * 2);

    dim3 blk(256);
    auto f2b = [&](const float* in, bf16* out, size_t n) {
        f2b_kernel<<<dim3((unsigned)((n + 255) / 256)), blk, 0, stream>>>(in, out, n);
    };
    auto ggrid = [](int M, int N) { return dim3((unsigned)(N / 128), (unsigned)(M / 128)); };

    // Weight conversions: all GEMM B operands stored [N,K] row-major bf16
    f2bT_kernel<<<dim3((256 * 512 + 255) / 256), blk, 0, stream>>>(gin_w1, w1tb, 256, 512);
    f2bT_kernel<<<dim3((512 * 256 + 255) / 256), blk, 0, stream>>>(gin_w2, w2tb, 512, 256);
    f2b(gru_wih, wihb, (size_t)2 * 768 * 256);   // already [N,K] (W^T multiply)
    f2b(gru_whh, whhb, (size_t)2 * 768 * 256);
    f2b(lstm_wih, lwihb, (size_t)1024 * 512);
    f2b(lstm_whh, lwhhb, (size_t)1024 * 256);

    // Node embeddings + hidden init
    embed_kernel<<<dim3(NN * 32 / 256), blk, 0, stream>>>(
        x_atoms, atom_emb, xf, hx0f, hx1f, hxb0, hxb1);

    // ----- 3 ENN layers -----
    for (int layer = 0; layer < 3; ++layer) {
        hipMemcpyAsync(aggf, xf, (size_t)NN * DH * 4, hipMemcpyDeviceToDevice, stream);
        message_kernel<<<dim3(NE * 32 / 256), blk, 0, stream>>>(xf, eidx, eattr, bond_emb, aggf);
        f2b(aggf, hbf, (size_t)NN * DH);
        // GIN MLP
        gemm_kernel<1><<<ggrid(NN, 512), blk, 0, stream>>>(
            hbf, w1tb, gin_b1, nullptr, tbf, NN, 512, 256);
        gemm_kernel<2><<<ggrid(NN, 256), blk, 0, stream>>>(
            tbf, w2tb, gin_b2, nullptr, xcb, NN, 256, 512);
        // GRU layer 0
        gemm_kernel<0><<<ggrid(NN, 768), blk, 0, stream>>>(
            xcb, wihb, gru_bih, gif, nullptr, NN, 768, 256);
        gemm_kernel<0><<<ggrid(NN, 768), blk, 0, stream>>>(
            hxb0, whhb, gru_bhh, ghf, nullptr, NN, 768, 256);
        gru_gate_kernel<<<dim3(NN * DH / 256), blk, 0, stream>>>(gif, ghf, hx0f, hxb0, nullptr);
        // GRU layer 1 (input = new h of layer 0); writes x for next layer
        gemm_kernel<0><<<ggrid(NN, 768), blk, 0, stream>>>(
            hxb0, wihb + (size_t)768 * 256, gru_bih + 768, gif, nullptr, NN, 768, 256);
        gemm_kernel<0><<<ggrid(NN, 768), blk, 0, stream>>>(
            hxb1, whhb + (size_t)768 * 256, gru_bhh + 768, ghf, nullptr, NN, 768, 256);
        gru_gate_kernel<<<dim3(NN * DH / 256), blk, 0, stream>>>(gif, ghf, hx1f, hxb1, xf);
    }

    // ----- Set2Set -----
    hipMemsetAsync(hsf, 0, (size_t)NG * DH * 4, stream);
    hipMemsetAsync(hsb, 0, (size_t)NG * DH * 2, stream);
    hipMemsetAsync(csf, 0, (size_t)NG * DH * 4, stream);
    hipMemsetAsync(qsf, 0, (size_t)NG * 512 * 4, stream);
    hipMemsetAsync(qsb, 0, (size_t)NG * 512 * 2, stream);
    hipMemsetAsync(cnt, 0, (size_t)NG * 4, stream);
    count_kernel<<<dim3(NN / 256), blk, 0, stream>>>(batch, cnt);
    scan_kernel<<<dim3(1), dim3(1), 0, stream>>>(cnt, off);

    for (int step = 0; step < 3; ++step) {
        gemm_kernel<0><<<ggrid(NG, 1024), blk, 0, stream>>>(
            qsb, lwihb, lstm_bih, g1f, nullptr, NG, 1024, 512);
        gemm_kernel<0><<<ggrid(NG, 1024), blk, 0, stream>>>(
            hsb, lwhhb, lstm_bhh, g2f, nullptr, NG, 1024, 256);
        lstm_gate_kernel<<<dim3(NG * DH / 256), blk, 0, stream>>>(g1f, g2f, csf, hsf, hsb);
        prod_kernel<<<dim3(NN * 32 / 256), blk, 0, stream>>>(hsf, xf, batch, prodv);
        attn_kernel<<<dim3(NG), blk, 0, stream>>>(off, prodv, xf, hsf, qsf, qsb);
    }

    final_kernel<<<dim3(NG * 32 / 256), blk, 0, stream>>>(qsf, fc_w, fc_b, outp);
}
